// Backbone_3143916060887
// MI455X (gfx1250) — matrix operand order repaired and checked by reading, not compiled, not run on hardware
//
#include <hip/hip_runtime.h>
#include <hip/hip_bf16.h>
#include <math.h>

// ---------------------------------------------------------------------------
// CDNA5 (gfx1250) conv backbone + MoE. All GEMMs on v_wmma_f32_16x16x32_bf16.
// Data is repacked once into NHWC / K-major bf16 so every WMMA fragment is
// filled with contiguous 16B/32B vector loads (no per-element address math).
// ---------------------------------------------------------------------------

typedef __attribute__((ext_vector_type(16))) __bf16 v16bf;
typedef __attribute__((ext_vector_type(8)))  __bf16 v8bf;
typedef __attribute__((ext_vector_type(8)))  float  v8f;

#define WMMA_BF16(a, b, c) \
    __builtin_amdgcn_wmma_f32_16x16x32_bf16(false, (a), false, (b), (short)0, (c), false, false)

// Build a 16x32 bf16 A-fragment from one contiguous 16-element vector load.
// Layout (matches B fills): lane group (lane>>4) selects K offset 0/16;
// elements 0..15 hold K = off..off+15.
__device__ __forceinline__ v16bf make_afrag(const __bf16* __restrict__ row, int hi8)
{
    return *(const v16bf*)(row + 2 * hi8);      // K = hi*16 .. +15
}

// ---------------------------------------------------------------------------
// Fused conv5x5(pad2)+bias+BN+ReLU+maxpool2x2, implicit GEMM on WMMA.
// Input  : NHWC bf16 [B,H,W,CIN_P]   (CIN_P = padded channels, mult of 32)
// Weights: bf16 [COUT, 25*CIN_P], K order k = (r*5+s)*CIN_P + c
// Output : NHWC bf16 [B,H/2,W/2,COUT]  or  NCHW f32 (last conv)
// One wave = 16 out-channels x 16 pooled pixels; 4 pool quadrants each run
// the full K reduction, then max-combine (same MAC count as unpooled conv).
// ---------------------------------------------------------------------------
template <int CIN_P, int COUT, int H, int W, bool NHWC_OUT>
__global__ __launch_bounds__(32)
void conv_bn_relu_pool_wmma(const __bf16* __restrict__ xn,
                            const __bf16* __restrict__ wp,
                            const float* __restrict__ bias,
                            const float* __restrict__ bng,
                            const float* __restrict__ bnb,
                            const float* __restrict__ bnm,
                            const float* __restrict__ bnv,
                            void* __restrict__ outv)
{
    constexpr int K  = 25 * CIN_P;
    constexpr int PH = H / 2, PW = W / 2;
    constexpr int NPIX = PH * PW;
    constexpr int TILES_P = (NPIX + 15) / 16;
    constexpr int TILES_C = COUT / 16;

    int wid = blockIdx.x;
    const int tc = wid % TILES_C; wid /= TILES_C;
    const int tp = wid % TILES_P; wid /= TILES_P;
    const int b  = wid;

    const int lane = threadIdx.x;
    const int lo   = lane & 15;
    const int hi   = (lane >> 4) & 1;
    const int hi8  = hi * 8;
    const int hi16 = hi * 16;

    const int pp    = tp * 16 + lo;
    const bool valid = (pp < NPIX);
    const int ppc = valid ? pp : 0;
    const int py = ppc / PW, px = ppc % PW;

    const __bf16* arow = wp + (long)(tc * 16 + lo) * K;   // A row = out channel

    float res[8];
#pragma unroll
    for (int v = 0; v < 8; ++v) res[v] = -1e30f;

#pragma unroll 1
    for (int q = 0; q < 4; ++q) {
        const int oy = py * 2 + (q >> 1);
        const int ox = px * 2 + (q & 1);
        v8f acc = {};

#pragma unroll 1
        for (int r = 0; r < 5; ++r) {
            const int ih = oy + r - 2;
            const bool rok = (ih >= 0) && (ih < H);
#pragma unroll 1
            for (int s = 0; s < 5; ++s) {
                const int iw = ox + s - 2;
                const bool ok = valid && rok && (iw >= 0) && (iw < W);
                const __bf16* bbase = xn + (((long)b * H + ih) * W + iw) * CIN_P;
                const int k0r = (r * 5 + s) * CIN_P;
#pragma unroll
                for (int c0 = 0; c0 < CIN_P; c0 += 32) {
                    v16bf bfrag = {};
                    if (ok) bfrag = *(const v16bf*)(bbase + c0 + hi16); // 16 contig ch
                    v16bf afrag = make_afrag(arow + k0r + c0, hi8);
                    acc = WMMA_BF16(afrag, bfrag, acc);
                }
            }
        }

        // bias + eval-BN + ReLU, max into pooled result
#pragma unroll
        for (int v = 0; v < 8; ++v) {
            const int ch = tc * 16 + hi8 + v;   // C/D layout: M = v + hi*8
            float val = acc[v] + bias[ch];
            float inv = rsqrtf(bnv[ch] + 1e-5f);
            val = (val - bnm[ch]) * (bng[ch] * inv) + bnb[ch];
            res[v] = fmaxf(res[v], fmaxf(val, 0.0f));
        }
    }

    if (valid) {
        if (NHWC_OUT) {
            __bf16* outb = (__bf16*)outv;
            v8bf o;
#pragma unroll
            for (int v = 0; v < 8; ++v) o[v] = (__bf16)res[v];
            *(v8bf*)(outb + (((long)b * PH + py) * PW + px) * COUT + tc * 16 + hi8) = o;
        } else {
            float* outf = (float*)outv;
#pragma unroll
            for (int v = 0; v < 8; ++v) {
                const int ch = tc * 16 + hi8 + v;
                outf[(((long)b * COUT + ch) * PH + py) * PW + px] = res[v];
            }
        }
    }
}

// ---------------------------------------------------------------------------
// Repack kernels (one pass each, bandwidth-trivial vs the conv reuse factor)
// ---------------------------------------------------------------------------

// x [B,CIN,H,W] f32 -> [B,H,W,CIN_P] bf16, zero-padded channels
__global__ void pack_x_nhwc(const float* __restrict__ x, __bf16* __restrict__ xn,
                            int B, int CIN, int CIN_P, int H, int W)
{
    long idx = (long)blockIdx.x * blockDim.x + threadIdx.x;
    long total = (long)B * H * W * CIN_P;
    if (idx >= total) return;
    int c = idx % CIN_P; long t = idx / CIN_P;
    int w = t % W; t /= W;
    int h = t % H; int b = t / H;
    float v = 0.0f;
    if (c < CIN) v = x[(((long)b * CIN + c) * H + h) * W + w];
    xn[idx] = (__bf16)v;
}

// w [COUT,CIN,5,5] f32 -> [COUT, 25*CIN_P] bf16 with k = rs*CIN_P + c
__global__ void pack_w(const float* __restrict__ w, __bf16* __restrict__ wp,
                       int COUT, int CIN, int CIN_P)
{
    long idx = (long)blockIdx.x * blockDim.x + threadIdx.x;
    long total = (long)COUT * 25 * CIN_P;
    if (idx >= total) return;
    int c  = idx % CIN_P; long t = idx / CIN_P;
    int rs = t % 25;  int m = t / 25;
    float v = 0.0f;
    if (c < CIN) v = w[((long)m * CIN + c) * 25 + rs];
    wp[idx] = (__bf16)v;
}

// ew1 [E,F,HID] f32 -> [E,HID,F] bf16 (column/K-major for B fragments)
__global__ void transpose_ew1(const float* __restrict__ ew, __bf16* __restrict__ ewt,
                              int E, int F, int HID)
{
    long idx = (long)blockIdx.x * blockDim.x + threadIdx.x;
    long total = (long)E * HID * F;
    if (idx >= total) return;
    int k = idx % F; long t = idx / F;
    int n = t % HID; int e = t / HID;
    ewt[idx] = (__bf16)ew[((long)e * F + k) * HID + n];
}

// ew2 [E,HID,D] f32 -> [E,D,HID] bf16
__global__ void transpose_ew2(const float* __restrict__ ew, __bf16* __restrict__ ewt,
                              int E, int HID, int D)
{
    long idx = (long)blockIdx.x * blockDim.x + threadIdx.x;
    long total = (long)E * D * HID;
    if (idx >= total) return;
    int k = idx % HID; long t = idx / HID;
    int d = t % D; int e = t / D;
    ewt[idx] = (__bf16)ew[((long)e * HID + k) * D + d];
}

// ---------------------------------------------------------------------------
// avgpool 2x2 on f32 NCHW [B,C,14,14] -> feat f32 + bf16 [B, C*7*7]
// (flatten order matches reference reshape: f = c*49 + py*7 + px)
// ---------------------------------------------------------------------------
__global__ void avgpool2_feat_kernel(const float* __restrict__ in,
                                     float* __restrict__ feat,
                                     __bf16* __restrict__ featb,
                                     int B, int C, int H, int W)
{
    int idx = blockIdx.x * blockDim.x + threadIdx.x;
    int PH = H / 2, PW = W / 2;
    int total = B * C * PH * PW;
    if (idx >= total) return;
    int px = idx % PW; int t = idx / PW;
    int py = t % PH;   t /= PH;
    int c  = t % C;    int b = t / C;
    const float* p = in + (((long)b * C + c) * H + py * 2) * W + px * 2;
    float v = 0.25f * (p[0] + p[1] + p[W] + p[W + 1]);
    feat[idx]  = v;
    featb[idx] = (__bf16)v;
}

// ---------------------------------------------------------------------------
// Router (tiny; f32 for fidelity)
// ---------------------------------------------------------------------------
__global__ void router_dense_kernel(const float* __restrict__ feat,
                                    const float* __restrict__ rw,
                                    const float* __restrict__ rb,
                                    float* __restrict__ logits,
                                    int B, int E, int F)
{
    int idx = blockIdx.x * blockDim.x + threadIdx.x;
    if (idx >= B * E) return;
    int e = idx % E, b = idx / E;
    const float* f = feat + (long)b * F;
    const float* wv = rw + (long)e * F;
    float s = 0.0f;
    for (int k = 0; k < F; ++k) s += f[k] * wv[k];
    logits[idx] = s + rb[e];
}

__global__ void router_top2_kernel(const float* __restrict__ logits,
                                   float* __restrict__ tw, int* __restrict__ tidx,
                                   int B)
{
    int b = blockIdx.x * blockDim.x + threadIdx.x;
    if (b >= B) return;
    float l[8];
    float mx = -1e30f;
    for (int e = 0; e < 8; ++e) { l[e] = logits[b * 8 + e]; mx = fmaxf(mx, l[e]); }
    float s = 0.0f;
    for (int e = 0; e < 8; ++e) { l[e] = __expf(l[e] - mx); s += l[e]; }
    for (int e = 0; e < 8; ++e) l[e] /= s;
    int i0 = 0;
    for (int e = 1; e < 8; ++e) if (l[e] > l[i0]) i0 = e;
    int i1 = (i0 == 0) ? 1 : 0;
    for (int e = 0; e < 8; ++e) if (e != i0 && l[e] > l[i1]) i1 = e;
    float w0 = l[i0], w1 = l[i1], sw = w0 + w1;
    tw[b * 2 + 0] = w0 / sw;  tw[b * 2 + 1] = w1 / sw;
    tidx[b * 2 + 0] = i0;     tidx[b * 2 + 1] = i1;
}

// ---------------------------------------------------------------------------
// Expert GEMM1: hid[b,e,h] = relu(feat . ew1[e,:,h] + eb1[e,h])
// One wave per (e, n-tile): covers all M=128 with 8 accumulators so each
// ew1t column streams from HBM exactly once; feat (1.6MB) stays L2-resident.
// ---------------------------------------------------------------------------
__global__ __launch_bounds__(32)
void expert_gemm1_wmma(const __bf16* __restrict__ featb,  // [B,F]
                       const __bf16* __restrict__ ew1t,   // [E,HID,F]
                       const float* __restrict__ eb1,     // [E,HID]
                       __bf16* __restrict__ hid,          // [B,E,HID]
                       int B, int E, int F, int HID)
{
    int wid = blockIdx.x;
    const int tn = wid % (HID / 16);
    const int e  = wid / (HID / 16);

    const int lane = threadIdx.x;
    const int lo   = lane & 15;
    const int hi8  = ((lane >> 4) & 1) * 8;
    const int hi16 = hi8 * 2;
    const int ncol = tn * 16 + lo;

    const __bf16* bcol = ew1t + ((long)e * HID + ncol) * F;

    v8f acc[8];
#pragma unroll
    for (int t = 0; t < 8; ++t) acc[t] = (v8f){};

#pragma unroll 1
    for (int k0 = 0; k0 < F; k0 += 32) {
        v16bf bfrag = *(const v16bf*)(bcol + k0 + hi16);
        __builtin_prefetch(bcol + k0 + 256, 0, 1);   // next chunks -> global_prefetch
#pragma unroll
        for (int tm = 0; tm < 8; ++tm) {
            const __bf16* arow = featb + (long)(tm * 16 + lo) * F + k0;
            v16bf afrag = make_afrag(arow, hi8);
            acc[tm] = WMMA_BF16(afrag, bfrag, acc[tm]);
        }
    }

    const float bv = eb1[(long)e * HID + ncol];
#pragma unroll
    for (int tm = 0; tm < 8; ++tm) {
#pragma unroll
        for (int v = 0; v < 8; ++v) {
            int mm = tm * 16 + hi8 + v;
            hid[((long)mm * E + e) * HID + ncol] = (__bf16)fmaxf(acc[tm][v] + bv, 0.0f);
        }
    }
}

// ---------------------------------------------------------------------------
// Expert GEMM2: all_out[b,e,d] = hid[b,e,:] . ew2[e,:,d] + eb2[e,d]
// ---------------------------------------------------------------------------
__global__ __launch_bounds__(32)
void expert_gemm2_wmma(const __bf16* __restrict__ hid,    // [B,E,HID]
                       const __bf16* __restrict__ ew2t,   // [E,D,HID]
                       const float* __restrict__ eb2,     // [E,D]
                       float* __restrict__ allo,          // [B,E,D]
                       int B, int E, int HID, int D)
{
    int wid = blockIdx.x;
    const int tn = wid % (D / 16);
    const int e  = wid / (D / 16);

    const int lane = threadIdx.x;
    const int lo   = lane & 15;
    const int hi8  = ((lane >> 4) & 1) * 8;
    const int hi16 = hi8 * 2;
    const int ncol = tn * 16 + lo;

    const __bf16* bcol = ew2t + ((long)e * D + ncol) * HID;

    v8f acc[8];
#pragma unroll
    for (int t = 0; t < 8; ++t) acc[t] = (v8f){};

#pragma unroll 1
    for (int k0 = 0; k0 < HID; k0 += 32) {
        v16bf bfrag = *(const v16bf*)(bcol + k0 + hi16);
#pragma unroll
        for (int tm = 0; tm < 8; ++tm) {
            const __bf16* arow = hid + ((long)(tm * 16 + lo) * E + e) * HID + k0;
            v16bf afrag = make_afrag(arow, hi8);
            acc[tm] = WMMA_BF16(afrag, bfrag, acc[tm]);
        }
    }

    const float bv = eb2[(long)e * D + ncol];
#pragma unroll
    for (int tm = 0; tm < 8; ++tm) {
#pragma unroll
        for (int v = 0; v < 8; ++v) {
            int mm = tm * 16 + hi8 + v;
            allo[((long)mm * E + e) * D + ncol] = acc[tm][v] + bv;
        }
    }
}

// ---------------------------------------------------------------------------
// Top-2 combine + L2 normalize. One block (256 threads = D) per batch row.
// ---------------------------------------------------------------------------
__global__ void combine_norm_kernel(const float* __restrict__ allo,
                                    const float* __restrict__ tw,
                                    const int* __restrict__ tidx,
                                    float* __restrict__ out,
                                    int E, int D)
{
    int b = blockIdx.x;
    int d = threadIdx.x;
    float v = 0.0f;
    for (int j = 0; j < 2; ++j) {
        int e = tidx[b * 2 + j];
        v += tw[b * 2 + j] * allo[((long)b * E + e) * D + d];
    }
    __shared__ float red[256];
    red[d] = v * v;
    __syncthreads();
    for (int s = 128; s > 0; s >>= 1) {
        if (d < s) red[d] += red[d + s];
        __syncthreads();
    }
    float norm = sqrtf(red[0]);
    out[(long)b * D + d] = v / fmaxf(norm, 1e-12f);
}

// ---------------------------------------------------------------------------
// Launch
// ---------------------------------------------------------------------------
extern "C" void kernel_launch(void* const* d_in, const int* in_sizes, int n_in,
                              void* d_out, int out_size, void* d_ws, size_t ws_size,
                              hipStream_t stream)
{
    const float* x    = (const float*)d_in[0];
    const float* c1w  = (const float*)d_in[1];
    const float* c1b  = (const float*)d_in[2];
    const float* g1   = (const float*)d_in[3];
    const float* b1   = (const float*)d_in[4];
    const float* m1   = (const float*)d_in[5];
    const float* v1   = (const float*)d_in[6];
    const float* c2w  = (const float*)d_in[7];
    const float* c2b  = (const float*)d_in[8];
    const float* g2   = (const float*)d_in[9];
    const float* b2   = (const float*)d_in[10];
    const float* m2   = (const float*)d_in[11];
    const float* v2   = (const float*)d_in[12];
    const float* c3w  = (const float*)d_in[13];
    const float* c3b  = (const float*)d_in[14];
    const float* g3   = (const float*)d_in[15];
    const float* b3   = (const float*)d_in[16];
    const float* m3   = (const float*)d_in[17];
    const float* v3   = (const float*)d_in[18];
    const float* rw   = (const float*)d_in[19];
    const float* rb   = (const float*)d_in[20];
    const float* ew1  = (const float*)d_in[21];
    const float* eb1  = (const float*)d_in[22];
    const float* ew2  = (const float*)d_in[23];
    const float* eb2  = (const float*)d_in[24];
    float* out = (float*)d_out;

    const int B = 128, E = 8, F = 6272, HID = 512, D = 256;

    // ---- workspace carve-up (256B aligned blocks) ----
    char* wsp = (char*)d_ws;
    auto carve = [&](size_t bytes) -> void* {
        void* p = (void*)wsp;
        wsp += (bytes + 255) & ~(size_t)255;
        return p;
    };
    __bf16* xn    = (__bf16*)carve((size_t)128 * 112 * 112 * 64 * 2); // conv1 in, Cin pad 64
    __bf16* w1p   = (__bf16*)carve((size_t)32 * 1600 * 2);
    __bf16* w2p   = (__bf16*)carve((size_t)64 * 800 * 2);
    __bf16* w3p   = (__bf16*)carve((size_t)128 * 1600 * 2);
    __bf16* h1    = (__bf16*)carve((size_t)128 * 56 * 56 * 32 * 2);   // NHWC bf16
    __bf16* h2    = (__bf16*)carve((size_t)128 * 28 * 28 * 64 * 2);   // NHWC bf16
    float*  h3    = (float*)carve((size_t)128 * 128 * 14 * 14 * 4);   // NCHW f32
    float*  feat  = (float*)carve((size_t)B * F * 4);
    __bf16* featb = (__bf16*)carve((size_t)B * F * 2);
    float*  logits= (float*)carve((size_t)B * E * 4);
    float*  tw    = (float*)carve((size_t)B * 2 * 4);
    int*    tidx  = (int*)carve((size_t)B * 2 * 4);
    __bf16* ew1t  = (__bf16*)carve((size_t)E * HID * F * 2);
    __bf16* ew2t  = (__bf16*)carve((size_t)E * D * HID * 2);
    __bf16* hid   = (__bf16*)carve((size_t)B * E * HID * 2);
    float*  allo  = (float*)carve((size_t)B * E * D * 4);

    // ---- repack passes ----
    {
        long tx = (long)128 * 112 * 112 * 64;
        pack_x_nhwc<<<(tx + 255) / 256, 256, 0, stream>>>(x, xn, 128, 36, 64, 112, 112);
        pack_w<<<((long)32 * 1600 + 255) / 256, 256, 0, stream>>>(c1w, w1p, 32, 36, 64);
        pack_w<<<((long)64 * 800 + 255) / 256, 256, 0, stream>>>(c2w, w2p, 64, 32, 32);
        pack_w<<<((long)128 * 1600 + 255) / 256, 256, 0, stream>>>(c3w, w3p, 128, 64, 64);
        long t1 = (long)E * HID * F;
        transpose_ew1<<<(t1 + 255) / 256, 256, 0, stream>>>(ew1, ew1t, E, F, HID);
        long t2 = (long)E * D * HID;
        transpose_ew2<<<(t2 + 255) / 256, 256, 0, stream>>>(ew2, ew2t, E, HID, D);
    }

    // ---- conv stack ----
    // conv1: 36(pad64)->32, 112x112, pooled 56x56 (196 pixel tiles, 2 ch tiles)
    conv_bn_relu_pool_wmma<64, 32, 112, 112, true>
        <<<B * 196 * 2, 32, 0, stream>>>(xn, w1p, c1b, g1, b1, m1, v1, (void*)h1);
    // conv2: 32->64, 56x56, pooled 28x28 (49 pixel tiles, 4 ch tiles)
    conv_bn_relu_pool_wmma<32, 64, 56, 56, true>
        <<<B * 49 * 4, 32, 0, stream>>>(h1, w2p, c2b, g2, b2, m2, v2, (void*)h2);
    // conv3: 64->128, 28x28, pooled 14x14 (13 ragged pixel tiles, 8 ch tiles)
    conv_bn_relu_pool_wmma<64, 128, 28, 28, false>
        <<<B * 13 * 8, 32, 0, stream>>>(h2, w3p, c3b, g3, b3, m3, v3, (void*)h3);

    // ---- avgpool -> feat (f32 + bf16) ----
    {
        int total = B * 128 * 7 * 7;
        avgpool2_feat_kernel<<<(total + 255) / 256, 256, 0, stream>>>(h3, feat, featb, B, 128, 14, 14);
    }

    // ---- router ----
    router_dense_kernel<<<(B * E + 255) / 256, 256, 0, stream>>>(feat, rw, rb, logits, B, E, F);
    router_top2_kernel<<<1, B, 0, stream>>>(logits, tw, tidx, B);

    // ---- experts (dense, as in reference) ----
    expert_gemm1_wmma<<<E * (HID / 16), 32, 0, stream>>>(featb, ew1t, eb1, hid, B, E, F, HID);
    expert_gemm2_wmma<<<E * (D / 16), 32, 0, stream>>>(hid, ew2t, eb2, allo, B, E, HID, D);

    // ---- combine + normalize ----
    combine_norm_kernel<<<B, D, 0, stream>>>(allo, tw, tidx, out, E, D);
}
